// RSSM_29841432772795
// MI455X (gfx1250) — matrix operand order repaired and checked by reading, not compiled, not run on hardware
//
#include <hip/hip_runtime.h>
#include <math.h>

typedef unsigned short u16;
typedef __bf16 v16bf __attribute__((ext_vector_type(16)));
typedef __bf16 v8bf  __attribute__((ext_vector_type(8)));
typedef __bf16 v4bf  __attribute__((ext_vector_type(4)));
typedef __bf16 v2bf  __attribute__((ext_vector_type(2)));
typedef float  v8f   __attribute__((ext_vector_type(8)));

#define B_SZ   1024
#define T_SZ   64
#define ACT    6
#define STOCH  30
#define DETER  200
#define HIDDEN 200
#define EMBED  1024
#define OUTW   580

// ---- packed weight layout in d_ws (bf16 elements) ----
#define NFRAG_WI   (13*2)
#define NFRAG_WG   (13*3*2*7)
#define NFRAG_WP1  (13*7)
#define NFRAG_WMS  (4*7)
#define NFRAG_WQ1  (13*39)
#define OFF_WI   0
#define OFF_WG   (OFF_WI  + NFRAG_WI*512)
#define OFF_WP1  (OFF_WG  + NFRAG_WG*512)
#define OFF_WP2  (OFF_WP1 + NFRAG_WP1*512)
#define OFF_WQ1  (OFF_WP2 + NFRAG_WMS*512)
#define OFF_WQ2  (OFF_WQ1 + NFRAG_WQ1*512)
#define OFF_WEND (OFF_WQ2 + NFRAG_WMS*512)
// bias region (fp32), placed after weights
#define BOFF_BI   0
#define BOFF_BIH  208
#define BOFF_BHH  832
#define BOFF_BP1  1456
#define BOFF_BP2  1664
#define BOFF_BQ1  1728
#define BOFF_BQ2  1936
#define BIAS_N    2000

// ---------------- device helpers ----------------
__device__ __forceinline__ v16bf load_a(const __bf16* buf, int ld, int k0, int lane) {
  // A-frag (16x32 bf16): lanes 0-15 row M=lane hold K={k0..k0+15};
  // lanes 16-31 row M=lane-16 hold K={k0+16..k0+31}.
  const __bf16* row = buf + (lane & 15) * ld + k0 + ((lane >> 4) << 4);
  v8bf lo = *(const v8bf*)(row);
  v8bf hi = *(const v8bf*)(row + 8);
  return __builtin_shufflevector(lo, hi, 0,1,2,3,4,5,6,7,8,9,10,11,12,13,14,15);
}
__device__ __forceinline__ v16bf load_b(const __bf16* base, int frag, int lane) {
  return *(const v16bf*)(base + (size_t)frag * 512 + lane * 16);
}
__device__ __forceinline__ v8f wmma_bf(v16bf a, v16bf b, v8f c) {
  return __builtin_amdgcn_wmma_f32_16x16x32_bf16(false, a, false, b, (short)0, c, false, false);
}
// Branch-free fast transcendentals (hardware v_exp/v_log/v_rcp; no exec-mask churn).
__device__ __forceinline__ float rcpf_hw(float x) { return __builtin_amdgcn_rcpf(x); }
__device__ __forceinline__ float eluf(float x) {
  // exact: for x>0 -> x; for x<=0 -> exp(x)-1 (and exp(x)-1 >= x always)
  return fmaxf(x, __expf(fminf(x, 0.f)) - 1.f);
}
__device__ __forceinline__ float sigf(float x) {
  return rcpf_hw(1.f + __expf(-x));
}
__device__ __forceinline__ float tanh_fast(float x) {
  float e = __expf(-2.f * fabsf(x));               // (0,1]
  float t = (1.f - e) * rcpf_hw(1.f + e);
  return copysignf(t, x);
}
__device__ __forceinline__ float softplusf(float x) {
  return fmaxf(x, 0.f) + __logf(1.f + __expf(-fabsf(x)));
}

// ---------------- weight packing kernels ----------------
// B-frag element map: lane<16 -> (N = nt*16+lane, K = kt*32+h); lane>=16 -> K += 16.
__global__ void pack_plain(const float* __restrict__ W, u16* __restrict__ dst,
                           int nTiles, int kTiles, int Nv, int Kv) {
  int id = blockIdx.x * 256 + threadIdx.x;
  if (id >= nTiles * kTiles * 512) return;
  int f = id >> 9, lane = (id >> 4) & 31, h = id & 15;
  int nt = f / kTiles, kt = f - nt * kTiles;
  int n = nt * 16 + (lane & 15);
  int k = kt * 32 + ((lane >> 4) << 4) + h;
  float v = (n < Nv && k < Kv) ? W[(size_t)n * Kv + k] : 0.f;
  reinterpret_cast<__bf16*>(dst)[id] = (__bf16)v;
}

__global__ void pack_gates(const float* __restrict__ Wih, const float* __restrict__ Whh,
                           u16* __restrict__ dst) {
  int id = blockIdx.x * 256 + threadIdx.x;
  if (id >= NFRAG_WG * 512) return;
  int f = id >> 9, lane = (id >> 4) & 31, h = id & 15;
  int kt = f % 7; int q = f / 7; int s = q & 1; q >>= 1; int g = q % 3; int nb = q / 3;
  int nl = nb * 16 + (lane & 15);
  int k = kt * 32 + ((lane >> 4) << 4) + h;
  const float* W = s ? Whh : Wih;
  float v = (nl < DETER && k < HIDDEN) ? W[(size_t)(g * DETER + nl) * HIDDEN + k] : 0.f;
  reinterpret_cast<__bf16*>(dst)[id] = (__bf16)v;
}

__global__ void pack_ms(const float* __restrict__ W, u16* __restrict__ dst) {
  int id = blockIdx.x * 256 + threadIdx.x;
  if (id >= NFRAG_WMS * 512) return;
  int f = id >> 9, lane = (id >> 4) & 31, h = id & 15;
  int kt = f % 7, nt = f / 7;
  int dloc = (nt & 1) * 16 + (lane & 15);
  int srow = (nt >= 2 ? STOCH : 0) + dloc;
  int k = kt * 32 + ((lane >> 4) << 4) + h;
  float v = (dloc < STOCH && k < HIDDEN) ? W[(size_t)srow * HIDDEN + k] : 0.f;
  reinterpret_cast<__bf16*>(dst)[id] = (__bf16)v;
}

__global__ void pack_bias(const float* bi, const float* bih, const float* bhh,
                          const float* bp1, const float* bp2, const float* bq1,
                          const float* bq2, float* dst) {
  int id = blockIdx.x * 256 + threadIdx.x;
  if (id >= BIAS_N) return;
  float v = 0.f;
  if (id < BOFF_BIH)      { int j = id;              if (j < HIDDEN) v = bi[j]; }
  else if (id < BOFF_BHH) { int j = id - BOFF_BIH; int nb = j/48, g = (j/16)%3, l = j&15;
                            int c = nb*16 + l; if (c < DETER) v = bih[g*DETER + c]; }
  else if (id < BOFF_BP1) { int j = id - BOFF_BHH; int nb = j/48, g = (j/16)%3, l = j&15;
                            int c = nb*16 + l; if (c < DETER) v = bhh[g*DETER + c]; }
  else if (id < BOFF_BP2) { int j = id - BOFF_BP1;   if (j < HIDDEN) v = bp1[j]; }
  else if (id < BOFF_BQ1) { int j = id - BOFF_BP2; int half = j >> 5, d = j & 31;
                            if (d < STOCH) v = bp2[half*STOCH + d]; }
  else if (id < BOFF_BQ2) { int j = id - BOFF_BQ1;   if (j < HIDDEN) v = bq1[j]; }
  else                    { int j = id - BOFF_BQ2; int half = j >> 5, d = j & 31;
                            if (d < STOCH) v = bq2[half*STOCH + d]; }
  dst[id] = v;
}

// ---------------- main persistent scan kernel ----------------
__global__ __launch_bounds__(256) void rssm_main(
    const float* __restrict__ embed, const float* __restrict__ action,
    const float* __restrict__ nprior, const float* __restrict__ npost,
    const u16* __restrict__ wpk_raw, const float* __restrict__ bpk,
    float* __restrict__ out)
{
  __shared__ __align__(16) __bf16 sBig[16 * 1248];   // [deter(200) | embed(1024) | pad(24)]
  __shared__ __align__(16) __bf16 sX  [16 * 224];    // x, then hp
  __shared__ __align__(16) __bf16 sHq [16 * 224];    // hq
  __shared__ float sDeterF[16 * DETER];              // fp32 deter state
  __shared__ float sStoch [16 * 32];                 // fp32 post stoch state
  __shared__ __align__(16) __bf16 sIn0[16 * 64];     // concat(stoch, action), K-pad 64

  const __bf16* wpk  = reinterpret_cast<const __bf16*>(wpk_raw);
  const __bf16* WiP  = wpk + OFF_WI;
  const __bf16* WgP  = wpk + OFF_WG;
  const __bf16* Wp1P = wpk + OFF_WP1;
  const __bf16* Wp2P = wpk + OFF_WP2;
  const __bf16* Wq1P = wpk + OFF_WQ1;
  const __bf16* Wq2P = wpk + OFF_WQ2;

  const int tid  = threadIdx.x;
  const int lane = tid & 31;
  const int w    = tid >> 5;
  const int b0   = blockIdx.x * 16;
  const int l15  = lane & 15;
  const int rb   = (lane < 16) ? 0 : 8;

  for (int i = tid; i < 16*1248; i += 256) sBig[i] = (__bf16)0.f;
  for (int i = tid; i < 16*224;  i += 256) { sX[i] = (__bf16)0.f; sHq[i] = (__bf16)0.f; }
  for (int i = tid; i < 16*DETER;i += 256) sDeterF[i] = 0.f;
  for (int i = tid; i < 16*32;   i += 256) sStoch[i] = 0.f;
  for (int i = tid; i < 16*64;   i += 256) sIn0[i] = (__bf16)0.f;
  __syncthreads();

  for (int t = 0; t < T_SZ; ++t) {
    // (a) stage embed tile (vectorized: float4 load -> 4x bf16 LDS store) + build concat(stoch, action)
    for (int i = tid; i < 16*(EMBED/4); i += 256) {
      int r = i >> 8;                      // EMBED/4 == 256 chunks per row
      int c4 = (i & 255) << 2;
      float4 v = *(const float4*)(embed + ((size_t)(b0+r)*T_SZ + t)*EMBED + c4);
      v4bf p; p[0] = (__bf16)v.x; p[1] = (__bf16)v.y; p[2] = (__bf16)v.z; p[3] = (__bf16)v.w;
      *(v4bf*)(&sBig[r*1248 + DETER + c4]) = p;
    }
    // prefetch next step's embed tile into cache (hides the only cold-HBM latency
    // of the serial scan); 128B lines, 32 lines per 4KB row, 512 lines total.
    if (t + 1 < T_SZ) {
      const float* nxt = embed + ((size_t)b0 * T_SZ + (t + 1)) * EMBED;
      for (int i = tid; i < 512; i += 256) {
        int r = i >> 5, line = i & 31;
        __builtin_prefetch(nxt + (size_t)r * T_SZ * EMBED + line * 32, 0, 1);
      }
    }
    for (int i = tid; i < 16*36; i += 256) {
      int r = i / 36, c = i - r*36;
      float v = (c < STOCH) ? sStoch[r*32 + c]
                            : action[((size_t)(b0+r)*T_SZ + t)*ACT + (c - STOCH)];
      sIn0[r*64 + c] = (__bf16)v;
    }
    __syncthreads();

    // (b) x = elu(concat(stoch,a) @ Wi^T + bi)
    for (int nt = w; nt < 13; nt += 8) {
      int col = nt*16 + l15;                 // <= 207, always in padded bias region
      float bia = bpk[BOFF_BI + col];        // issue early; completes under the GEMM
      v8f acc = {};
      #pragma unroll
      for (int kt = 0; kt < 2; ++kt) {
        v16bf a = load_a(sIn0, 64, kt*32, lane);
        acc = wmma_bf(a, load_b(WiP, nt*2 + kt, lane), acc);
      }
      if (col < HIDDEN) {
        #pragma unroll
        for (int r = 0; r < 8; ++r)
          sX[(rb + r)*224 + col] = (__bf16)eluf(acc[r] + bia);
      }
    }
    __syncthreads();

    // (c) GRU: 6 accumulators per column-block, gates fully in registers
    for (int nb = w; nb < 13; nb += 8) {
      float bir = bpk[BOFF_BIH + nb*48 +  0 + l15];
      float biz = bpk[BOFF_BIH + nb*48 + 16 + l15];
      float bin = bpk[BOFF_BIH + nb*48 + 32 + l15];
      float bhr = bpk[BOFF_BHH + nb*48 +  0 + l15];
      float bhz = bpk[BOFF_BHH + nb*48 + 16 + l15];
      float bhn = bpk[BOFF_BHH + nb*48 + 32 + l15];
      v8f air = {}, aiz = {}, ain = {}, ahr = {}, ahz = {}, ahn = {};
      for (int kt = 0; kt < 7; ++kt) {
        v16bf ax = load_a(sX,   224,  kt*32, lane);
        v16bf ad = load_a(sBig, 1248, kt*32, lane);   // old deter (embed cols zero-weighted)
        air = wmma_bf(ax, load_b(WgP, ((nb*3+0)*2+0)*7 + kt, lane), air);
        ahr = wmma_bf(ad, load_b(WgP, ((nb*3+0)*2+1)*7 + kt, lane), ahr);
        aiz = wmma_bf(ax, load_b(WgP, ((nb*3+1)*2+0)*7 + kt, lane), aiz);
        ahz = wmma_bf(ad, load_b(WgP, ((nb*3+1)*2+1)*7 + kt, lane), ahz);
        ain = wmma_bf(ax, load_b(WgP, ((nb*3+2)*2+0)*7 + kt, lane), ain);
        ahn = wmma_bf(ad, load_b(WgP, ((nb*3+2)*2+1)*7 + kt, lane), ahn);
      }
      int col = nb*16 + l15;
      if (col < DETER) {
        #pragma unroll
        for (int r = 0; r < 8; ++r) {
          float rg = sigf(air[r] + bir + ahr[r] + bhr);
          float zg = sigf(aiz[r] + biz + ahz[r] + bhz);
          float ng = tanh_fast(ain[r] + bin + rg * (ahn[r] + bhn));
          int idx = (rb + r)*DETER + col;
          float hold = sDeterF[idx];
          sDeterF[idx] = (1.f - zg)*ng + zg*hold;
        }
      }
    }
    __syncthreads();

    // (d) publish deter_new: bf16 pair to LDS, float2 to output (x2), vectorized
    for (int i = tid; i < 16*(DETER/2); i += 256) {
      int r = i / 100, c = (i - r*100) * 2;
      float v0 = sDeterF[r*DETER + c];
      float v1 = sDeterF[r*DETER + c + 1];
      v2bf pb; pb[0] = (__bf16)v0; pb[1] = (__bf16)v1;
      *(v2bf*)(&sBig[r*1248 + c]) = pb;
      size_t ob = ((size_t)(b0+r)*T_SZ + t)*OUTW;
      float2 pf; pf.x = v0; pf.y = v1;
      *(float2*)(out + ob +  90 + c) = pf;
      *(float2*)(out + ob + 380 + c) = pf;
    }
    __syncthreads();

    // (e) hp = elu(deter @ Wp1^T + bp1); hq = elu(concat(deter,embed) @ Wq1^T + bq1)
    for (int nt = w; nt < 13; nt += 8) {
      int col = nt*16 + l15;
      float bia = bpk[BOFF_BP1 + col];
      v8f acc = {};
      for (int kt = 0; kt < 7; ++kt) {
        v16bf a = load_a(sBig, 1248, kt*32, lane);
        acc = wmma_bf(a, load_b(Wp1P, nt*7 + kt, lane), acc);
      }
      if (col < HIDDEN) {
        #pragma unroll
        for (int r = 0; r < 8; ++r)
          sX[(rb + r)*224 + col] = (__bf16)eluf(acc[r] + bia);
      }
    }
    for (int nt = w; nt < 13; nt += 8) {
      int col = nt*16 + l15;
      float bia = bpk[BOFF_BQ1 + col];
      v8f acc = {};
      for (int kt = 0; kt < 39; ++kt) {
        v16bf a = load_a(sBig, 1248, kt*32, lane);
        acc = wmma_bf(a, load_b(Wq1P, nt*39 + kt, lane), acc);
      }
      if (col < HIDDEN) {
        #pragma unroll
        for (int r = 0; r < 8; ++r)
          sHq[(rb + r)*224 + col] = (__bf16)eluf(acc[r] + bia);
      }
    }
    __syncthreads();

    // (f) heads: waves 0-1 prior, waves 2-3 posterior (mean+std tiles paired in-register)
    if (w < 4) {
      const int  j     = w & 1;
      const bool prior = (w < 2);
      const __bf16* Wh  = prior ? Wp2P : Wq2P;
      const __bf16* src = prior ? sX   : sHq;
      const int  bofs   = prior ? BOFF_BP2 : BOFF_BQ2;
      float bm = bpk[bofs + j*16 + l15];
      float bs = bpk[bofs + 32 + j*16 + l15];
      v8f am = {}, as = {};
      for (int kt = 0; kt < 7; ++kt) {
        v16bf a = load_a(src, 224, kt*32, lane);
        am = wmma_bf(a, load_b(Wh, j*7 + kt,     lane), am);
        as = wmma_bf(a, load_b(Wh, (j+2)*7 + kt, lane), as);
      }
      int d = j*16 + l15;
      if (d < STOCH) {
        const float* nz = prior ? nprior : npost;
        #pragma unroll
        for (int r = 0; r < 8; ++r) {
          int row = rb + r;
          float m  = am[r] + bm;
          float s  = softplusf(as[r] + bs) + 0.1f;
          float n  = nz[((size_t)(b0+row)*T_SZ + t)*STOCH + d];
          float st = n*s + m;
          size_t ob = ((size_t)(b0+row)*T_SZ + t)*OUTW;
          if (prior) { out[ob + 290 + d] = m; out[ob + 320 + d] = s; out[ob + 350 + d] = st; }
          else       { out[ob +   0 + d] = m; out[ob +  30 + d] = s; out[ob +  60 + d] = st;
                       sStoch[row*32 + d] = st; }
        }
      }
    }
    __syncthreads();
  }
}

// ---------------- launcher ----------------
extern "C" void kernel_launch(void* const* d_in, const int* in_sizes, int n_in,
                              void* d_out, int out_size, void* d_ws, size_t ws_size,
                              hipStream_t stream) {
  (void)in_sizes; (void)n_in; (void)out_size; (void)ws_size;
  const float* embed  = (const float*)d_in[0];
  const float* action = (const float*)d_in[1];
  const float* nprior = (const float*)d_in[2];
  const float* npost  = (const float*)d_in[3];
  const float* Wi  = (const float*)d_in[4];
  const float* bi  = (const float*)d_in[5];
  const float* Wih = (const float*)d_in[6];
  const float* Whh = (const float*)d_in[7];
  const float* bih = (const float*)d_in[8];
  const float* bhh = (const float*)d_in[9];
  const float* Wp1 = (const float*)d_in[10];
  const float* bp1 = (const float*)d_in[11];
  const float* Wp2 = (const float*)d_in[12];
  const float* bp2 = (const float*)d_in[13];
  const float* Wq1 = (const float*)d_in[14];
  const float* bq1 = (const float*)d_in[15];
  const float* Wq2 = (const float*)d_in[16];
  const float* bq2 = (const float*)d_in[17];

  u16*   ws  = (u16*)d_ws;
  float* bws = (float*)((char*)d_ws + (size_t)OFF_WEND * 2);

  auto blocks = [](int total) { return (total + 255) / 256; };
  pack_plain<<<blocks(NFRAG_WI *512), 256, 0, stream>>>(Wi,  ws + OFF_WI,  13,  2, HIDDEN, STOCH + ACT);
  pack_gates<<<blocks(NFRAG_WG *512), 256, 0, stream>>>(Wih, Whh, ws + OFF_WG);
  pack_plain<<<blocks(NFRAG_WP1*512), 256, 0, stream>>>(Wp1, ws + OFF_WP1, 13,  7, HIDDEN, DETER);
  pack_ms   <<<blocks(NFRAG_WMS*512), 256, 0, stream>>>(Wp2, ws + OFF_WP2);
  pack_plain<<<blocks(NFRAG_WQ1*512), 256, 0, stream>>>(Wq1, ws + OFF_WQ1, 13, 39, HIDDEN, DETER + EMBED);
  pack_ms   <<<blocks(NFRAG_WMS*512), 256, 0, stream>>>(Wq2, ws + OFF_WQ2);
  pack_bias <<<blocks(BIAS_N),        256, 0, stream>>>(bi, bih, bhh, bp1, bp2, bq1, bq2, bws);

  rssm_main<<<64, 256, 0, stream>>>(embed, action, nprior, npost, ws, bws, (float*)d_out);
}
